// FPN_Net_3135326126384
// MI455X (gfx1250) — matrix operand order repaired and checked by reading, not compiled, not run on hardware
//
#include <hip/hip_runtime.h>
#include <hip/hip_bf16.h>

// ============================================================================
// FPN / sparse-U-Net forward on gfx1250 (MI455X).
// - All convs via v_wmma_f32_16x16x32_bf16 (bf16 A/B, f32 accumulate).
// - Activations stored bf16 (pre-masked) by the BN kernel.
// - Weights prepacked to exact B-fragment layout; the 8 waves of each block
//   share one nTile, so B fragments are staged once per block into LDS
//   (ds_load_b128 in the hot loop), cutting global B traffic 8x.
// - global_prefetch_b8 (__builtin_prefetch) hides A-gather latency.
//
// WMMA layouts (CDNA5 ISA, wave32):
//   A (16x32 bf16): lane L: m=L&15, khalf=L>>4; element e -> K = khalf*8 +
//                   (e<8 ? e : e+8)  == two 8-channel contiguous runs.
//   B (32x16 bf16): lane L: n=L&15, khalf=L>>4; element e -> K = khalf*8 + (e<8 ? e : e+8), same as A.
//   C/D (16x16 f32): VGPR r, lane L: row = r + (L>=16?8:0), col = L&15.
// ============================================================================

typedef __attribute__((ext_vector_type(16))) __bf16 v16bf;
typedef __attribute__((ext_vector_type(8)))  float  v8f;

struct alignas(16) U4 { unsigned x, y, z, w; };
struct U8 { U4 lo, hi; };

__device__ __forceinline__ __bf16 f2bf(float f) {
  union { float f; unsigned u; } in; in.f = f;
  unsigned u = in.u;
  u += 0x7FFFu + ((u >> 16) & 1u);           // round-to-nearest-even
  unsigned short h = (unsigned short)(u >> 16);
  return __builtin_bit_cast(__bf16, h);
}

__device__ __forceinline__ __bf16 bfzero() {
  return __builtin_bit_cast(__bf16, (unsigned short)0);
}

// ---------------------------------------------------------------------------
// Weight prepack: fp32 [taps, Cin, Cout] -> bf16 B-fragment blocks.
// Block (tap, kc, nT) holds 512 bf16 (= 64 U4): lane-major, 16 per lane.
// Out-of-range K rows / N cols zero-padded (handles Cin=3, Cout=20).
// ---------------------------------------------------------------------------
__global__ void prepack_w_k(const float* w, __bf16* wp, int taps, int Cin,
                            int Cout) {
  int kchunks = (Cin + 31) >> 5, nTiles = (Cout + 15) >> 4;
  long total = (long)taps * kchunks * nTiles * 512;
  long i = (long)blockIdx.x * blockDim.x + threadIdx.x;
  if (i >= total) return;
  int e = (int)(i & 15);
  int lane = (int)((i >> 4) & 31);
  long blk = i >> 9;
  int nT = (int)(blk % nTiles); long t2 = blk / nTiles;
  int kc = (int)(t2 % kchunks); int tap = (int)(t2 / kchunks);
  int khalf = lane >> 4, n = lane & 15;
  int row = kc * 32 + khalf * 8 + (e < 8 ? e : e + 8);
  int nn = nT * 16 + n;
  float v = (row < Cin && nn < Cout) ? w[((long)tap * Cin + row) * Cout + nn]
                                     : 0.f;
  wp[i] = f2bf(v);
}

// ---------------------------------------------------------------------------
// Generic implicit-GEMM conv (3^3 sconv pad=1, 2^3 s2 down, 1^3 linear+bias).
// x is bf16, pre-masked. Fast vector A path when Cin % 32 == 0.
// B fragments staged in dynamic LDS per block (all waves share nTile).
// ---------------------------------------------------------------------------
struct ConvP {
  const __bf16* x;       // [Vin, Cin] bf16 (pre-masked)
  const __bf16* wp;      // prepacked fragments
  const float* outMask;  // [Vout]
  const float* bias;     // [Cout] or null
  float* out;            // [Vout, outStride]
  int Bn, inX, inY, inZ;
  int outX, outY, outZ;
  int Cin, Cout, kdim, stride, pad;
  int outStride, outOff;
};

__global__ __launch_bounds__(256) void conv_wmma_k(ConvP p) {
  extern __shared__ U4 smemB[];     // [ktaps * kchunks * 64] U4
  const int lane = threadIdx.x & 31;
  const int wavesPerBlk = blockDim.x >> 5;
  const int wave = blockIdx.x * wavesPerBlk + (threadIdx.x >> 5);
  const long Vout = (long)p.Bn * p.outX * p.outY * p.outZ;
  const int voxTiles = (int)(Vout >> 4);
  const int nTiles = (p.Cout + 15) >> 4;
  const int kchunks = (p.Cin + 31) >> 5;
  const int ktaps = p.kdim * p.kdim * p.kdim;

  // ---- cooperative stage of this block's B fragments (uniform nTile) ----
  {
    const int nTileBlk = (blockIdx.x * wavesPerBlk) / voxTiles;
    const int fragU4 = ktaps * kchunks * 64;
    const U4* wp4 = (const U4*)p.wp;
    for (int i = threadIdx.x; i < fragU4; i += blockDim.x) {
      int tapkc = i >> 6, j = i & 63;
      smemB[i] = wp4[((long)tapkc * nTiles + nTileBlk) * 64 + j];
    }
  }
  __syncthreads();

  if (wave >= voxTiles * nTiles) return;
  const int vTile = wave % voxTiles;
  const int nTile = wave / voxTiles;
  const int khalf = lane >> 4;
  const int nn = nTile * 16 + (lane & 15);
  const bool fastA = (p.Cin & 31) == 0;

  int vox = vTile * 16 + (lane & 15);
  int oz = vox % p.outZ; int t = vox / p.outZ;
  int oy = t % p.outY;   t /= p.outY;
  int ox = t % p.outX;   int b = t / p.outX;
  const int ix0 = ox * p.stride - p.pad;
  const int iy0 = oy * p.stride - p.pad;
  const int iz0 = oz * p.stride - p.pad;

  v8f acc = {};
  const U4 zer = {0u, 0u, 0u, 0u};

  for (int tap = 0; tap < ktaps; ++tap) {
    int tz = tap % p.kdim, tt = tap / p.kdim;
    int ty = tt % p.kdim,  tx = tt / p.kdim;
    int ix = ix0 + tx, iy = iy0 + ty, iz = iz0 + tz;
    bool valid = (unsigned)ix < (unsigned)p.inX &&
                 (unsigned)iy < (unsigned)p.inY &&
                 (unsigned)iz < (unsigned)p.inZ;
    long inVox = (((long)b * p.inX + ix) * p.inY + iy) * p.inZ + iz;
    const __bf16* xrow = p.x + inVox * (long)p.Cin;

    // prefetch next tap's A row (global_prefetch_b8)
    if (tap + 1 < ktaps) {
      int t2z = (tap + 1) % p.kdim, t2t = (tap + 1) / p.kdim;
      int jx = ix0 + t2t / p.kdim, jy = iy0 + t2t % p.kdim, jz = iz0 + t2z;
      if ((unsigned)jx < (unsigned)p.inX && (unsigned)jy < (unsigned)p.inY &&
          (unsigned)jz < (unsigned)p.inZ) {
        long nv = (((long)b * p.inX + jx) * p.inY + jy) * p.inZ + jz;
        __builtin_prefetch(p.x + nv * (long)p.Cin, 0, 0);
      }
    }

    for (int kc = 0; kc < kchunks; ++kc) {
      v16bf a;
      if (fastA) {
        U4 lo = zer, hi = zer;
        if (valid) {
          const U4* ap = (const U4*)(xrow + kc * 32 + khalf * 8);
          lo = ap[0];                 // channels ch0 .. ch0+7
          hi = ap[2];                 // channels ch0+16 .. ch0+23
        }
        U8 t8 = {lo, hi};
        a = __builtin_bit_cast(v16bf, t8);
      } else {
#pragma unroll
        for (int e = 0; e < 16; ++e) {
          int ch = kc * 32 + khalf * 8 + (e < 8 ? e : e + 8);
          a[e] = (valid && ch < p.Cin) ? xrow[ch] : bfzero();
        }
      }
      const U4* bp = smemB + (tap * kchunks + kc) * 64 + lane * 2;
      U8 b8 = {bp[0], bp[1]};
      v16bf bm = __builtin_bit_cast(v16bf, b8);
      acc = __builtin_amdgcn_wmma_f32_16x16x32_bf16(
          false, a, false, bm, (short)0, acc, false, false);
    }
  }

  if (nn < p.Cout) {
    float bias = p.bias ? p.bias[nn] : 0.f;
#pragma unroll
    for (int r = 0; r < 8; ++r) {
      long vr = (long)vTile * 16 + r + khalf * 8;
      p.out[vr * p.outStride + p.outOff + nn] = (acc[r] + bias) * p.outMask[vr];
    }
  }
}

// ---------------------------------------------------------------------------
// Transpose conv 2^3 s2: tiled by (tap, 16 coarse voxels), scatter to fine.
// All waves in a block share (tap, nTile) -> stage B in LDS.
// ---------------------------------------------------------------------------
struct UpP {
  const __bf16* x;       // [Vc, Cin] bf16 (pre-masked)
  const __bf16* wp;      // prepacked fragments (8 taps)
  const float* outMask;  // [Vf]
  float* out;            // [Vf, outStride]
  int Bn, cX, cY, cZ, fX, fY, fZ;
  int Cin, Cout, outStride, outOff;
};

__global__ __launch_bounds__(256) void upconv_wmma_k(UpP p) {
  extern __shared__ U4 smemB[];     // [kchunks * 64] U4
  const int lane = threadIdx.x & 31;
  const int wavesPerBlk = blockDim.x >> 5;
  const int wave = blockIdx.x * wavesPerBlk + (threadIdx.x >> 5);
  const long Vc = (long)p.Bn * p.cX * p.cY * p.cZ;
  const int cTiles = (int)(Vc >> 4);
  const int nTiles = (p.Cout + 15) >> 4;
  const int kchunks = (p.Cin + 31) >> 5;

  // ---- stage B fragments for this block's uniform (tap, nTile) ----
  int wave0 = blockIdx.x * wavesPerBlk;
  const int tapBlk = wave0 / (cTiles * nTiles);
  const int nTileBlk = (wave0 % (cTiles * nTiles)) / cTiles;
  {
    const int fragU4 = kchunks * 64;
    const U4* wp4 = (const U4*)p.wp;
    for (int i = threadIdx.x; i < fragU4; i += blockDim.x) {
      int kc = i >> 6, j = i & 63;
      smemB[i] = wp4[(((long)tapBlk * kchunks + kc) * nTiles + nTileBlk) * 64 + j];
    }
  }
  __syncthreads();

  if (wave >= 8 * cTiles * nTiles) return;
  const int tap = wave / (cTiles * nTiles);
  const int rem = wave % (cTiles * nTiles);
  const int cTile = rem % cTiles;
  const int tz = tap & 1, ty = (tap >> 1) & 1, tx = (tap >> 2) & 1;
  const int khalf = lane >> 4;
  const int nn = ((wave / cTiles) % nTiles) * 16 + (lane & 15);

  const __bf16* xrow = p.x + ((long)cTile * 16 + (lane & 15)) * p.Cin;

  v8f acc = {};
  for (int kc = 0; kc < kchunks; ++kc) {
    const U4* ap = (const U4*)(xrow + kc * 32 + khalf * 8);
    U8 a8 = {ap[0], ap[2]};
    v16bf a = __builtin_bit_cast(v16bf, a8);
    const U4* bp = smemB + kc * 64 + lane * 2;
    U8 b8 = {bp[0], bp[1]};
    v16bf bm = __builtin_bit_cast(v16bf, b8);
    acc = __builtin_amdgcn_wmma_f32_16x16x32_bf16(
        false, a, false, bm, (short)0, acc, false, false);
  }

  if (nn < p.Cout) {
#pragma unroll
    for (int r = 0; r < 8; ++r) {
      long cv = (long)cTile * 16 + r + khalf * 8;
      int cz = (int)(cv % p.cZ); long t = cv / p.cZ;
      int cy = (int)(t % p.cY);  t /= p.cY;
      int cx = (int)(t % p.cX);  int b = (int)(t / p.cX);
      long f = (((long)b * p.fX + 2 * cx + tx) * p.fY + 2 * cy + ty) * p.fZ
               + 2 * cz + tz;
      p.out[f * p.outStride + p.outOff + nn] = acc[r] * p.outMask[f];
    }
  }
}

// ---------------------------------------------------------------------------
// Masked BatchNorm stats (sum, sumsq per channel; mask count in stats[2C]).
// ---------------------------------------------------------------------------
__global__ __launch_bounds__(256) void bn_stats_k(const float* x, const float* m,
                                                  float* stats, int C, int nVox) {
  int c = blockIdx.y;
  float s1 = 0.f, s2 = 0.f, sc = 0.f;
  for (int v = blockIdx.x * blockDim.x + threadIdx.x; v < nVox;
       v += gridDim.x * blockDim.x) {
    float mm = m[v];
    if (mm != 0.f) {
      float val = x[(long)v * C + c];
      s1 += val; s2 += val * val; sc += mm;
    }
  }
  __shared__ float a1[256], a2[256], a3[256];
  int t = threadIdx.x;
  a1[t] = s1; a2[t] = s2; a3[t] = sc;
  __syncthreads();
  for (int off = 128; off > 0; off >>= 1) {
    if (t < off) { a1[t] += a1[t + off]; a2[t] += a2[t + off]; a3[t] += a3[t + off]; }
    __syncthreads();
  }
  if (t == 0) {
    atomicAdd(&stats[c], a1[0]);
    atomicAdd(&stats[C + c], a2[0]);
    if (c == 0) atomicAdd(&stats[2 * C], a3[0]);
  }
}

// BN apply + ReLU + mask; emits bf16 (next conv's A operand).
__global__ void bn_apply_k(const float* x, const float* m, const float* g,
                           const float* b, const float* stats, int C, long total,
                           __bf16* out) {
  long i = (long)blockIdx.x * blockDim.x + threadIdx.x;
  if (i >= total) return;
  long v = i / C; int c = (int)(i % C);
  float cnt  = fmaxf(stats[2 * C], 1.f);
  float mean = stats[c] / cnt;
  float var  = stats[C + c] / cnt - mean * mean;
  float y = (x[i] - mean) * rsqrtf(var + 1e-4f) * g[c] + b[c];
  y = y > 0.f ? y : 0.f;              // leak = 0 -> ReLU
  out[i] = f2bf(y * m[v]);
}

// ---------------------------------------------------------------------------
// Elementwise helpers.
// ---------------------------------------------------------------------------
__global__ void mask2float_k(const unsigned char* mb, float* mf, long n) {
  long i = (long)blockIdx.x * blockDim.x + threadIdx.x;
  if (i < n) mf[i] = mb[i] ? 1.f : 0.f;
}

__global__ void feats2bf_k(const float* feats, const float* m, __bf16* out,
                           long V, int C) {
  long i = (long)blockIdx.x * blockDim.x + threadIdx.x;
  if (i >= V * C) return;
  out[i] = f2bf(feats[i] * m[i / C]);
}

__global__ void pool_mask_k(const float* fm, float* cm, int Bn, int cX, int cY,
                            int cZ) {
  long i = (long)blockIdx.x * blockDim.x + threadIdx.x;
  long Vc = (long)Bn * cX * cY * cZ;
  if (i >= Vc) return;
  int cz = (int)(i % cZ); long t = i / cZ;
  int cy = (int)(t % cY); t /= cY;
  int cx = (int)(t % cX); int b = (int)(t / cX);
  int fX = 2 * cX, fY = 2 * cY, fZ = 2 * cZ;
  float mx = 0.f;
  for (int dx = 0; dx < 2; ++dx)
    for (int dy = 0; dy < 2; ++dy)
      for (int dz = 0; dz < 2; ++dz) {
        long f = (((long)b * fX + 2 * cx + dx) * fY + 2 * cy + dy) * fZ
                 + 2 * cz + dz;
        mx = fmaxf(mx, fm[f]);
      }
  cm[i] = mx;
}

__global__ void concat_copy_k(const float* src, float* dst, long V, int C,
                              int stride, int off) {
  long i = (long)blockIdx.x * blockDim.x + threadIdx.x;
  if (i >= V * C) return;
  long v = i / C; int c = (int)(i % C);
  dst[v * stride + off + c] = src[i];
}

// ---------------------------------------------------------------------------
// Host orchestration.
// ---------------------------------------------------------------------------
extern "C" void kernel_launch(void* const* d_in, const int* in_sizes, int n_in,
                              void* d_out, int out_size, void* d_ws, size_t ws_size,
                              hipStream_t stream) {
  (void)in_sizes; (void)n_in; (void)out_size; (void)ws_size;

  const float* feats = (const float*)d_in[0];
  const unsigned char* maskB = (const unsigned char*)d_in[1];
  auto F = [&](int i) { return (const float*)d_in[i]; };

  const int Bn = 2;
  const int S0 = 64, S1 = 32, S2 = 16, S3 = 8;
  const long V0 = (long)Bn * S0 * S0 * S0;
  const long V1 = (long)Bn * S1 * S1 * S1;
  const long V2 = (long)Bn * S2 * S2 * S2;
  const long V3 = (long)Bn * S3 * S3 * S3;

  float* p = (float*)d_ws;
  float* m0 = p;    p += V0;
  float* m1 = p;    p += V1;
  float* m2 = p;    p += V2;
  float* m3 = p;    p += V3;
  float* stats = p; p += 512;
  float* bufF = p;  p += V0 * 64;            // f32 conv-output / concat buffer
  float* dn0 = p;   p += V0 * 32;
  float* dn1 = p;   p += V1 * 64;
  float* dn2 = p;   p += V2 * 96;
  float* dn3 = p;   p += V3 * 128;
  __bf16* xbf = (__bf16*)p;  p += V0 * 32;   // bf16 activations (V0*64 bf16)
  __bf16* fbf = (__bf16*)p;  p += (V0 * 3 + 1) / 2 + 8;  // feats*m bf16
  __bf16* wpk = (__bf16*)p;                  // prepacked weights (<=512K bf16)

  auto prepack = [&](const float* w, int taps, int Cin, int Cout) {
    int kchunks = (Cin + 31) >> 5, nTiles = (Cout + 15) >> 4;
    long total = (long)taps * kchunks * nTiles * 512;
    prepack_w_k<<<(int)((total + 255) / 256), 256, 0, stream>>>(w, wpk, taps,
                                                               Cin, Cout);
  };

  auto conv = [&](const __bf16* x, const float* w, const float* outM,
                  const float* bias, float* out, int iS, int oS, int Cin,
                  int Cout, int kd, int st, int pd, int ostr, int ooff) {
    prepack(w, kd * kd * kd, Cin, Cout);
    ConvP cp;
    cp.x = x; cp.wp = wpk; cp.outMask = outM; cp.bias = bias; cp.out = out;
    cp.Bn = Bn; cp.inX = iS; cp.inY = iS; cp.inZ = iS;
    cp.outX = oS; cp.outY = oS; cp.outZ = oS;
    cp.Cin = Cin; cp.Cout = Cout; cp.kdim = kd; cp.stride = st; cp.pad = pd;
    cp.outStride = ostr; cp.outOff = ooff;
    long Vout = (long)Bn * oS * oS * oS;
    long tiles = (Vout >> 4) * ((Cout + 15) >> 4);
    int kchunks = (Cin + 31) >> 5;
    size_t shmem = (size_t)kd * kd * kd * kchunks * 1024;   // B fragments
    conv_wmma_k<<<(int)((tiles + 7) / 8), 256, shmem, stream>>>(cp);
  };

  auto upconv = [&](const __bf16* x, const float* w, const float* outM,
                    float* out, int cS, int fS, int Cin, int Cout, int ostr) {
    prepack(w, 8, Cin, Cout);
    UpP up;
    up.x = x; up.wp = wpk; up.outMask = outM; up.out = out;
    up.Bn = Bn; up.cX = cS; up.cY = cS; up.cZ = cS;
    up.fX = fS; up.fY = fS; up.fZ = fS;
    up.Cin = Cin; up.Cout = Cout; up.outStride = ostr; up.outOff = 0;
    long Vc = (long)Bn * cS * cS * cS;
    long tiles = 8 * (Vc >> 4) * ((Cout + 15) >> 4);
    int kchunks = (Cin + 31) >> 5;
    size_t shmem = (size_t)kchunks * 1024;
    upconv_wmma_k<<<(int)((tiles + 7) / 8), 256, shmem, stream>>>(up);
  };

  auto bn = [&](const float* x, const float* mask, const float* g,
                const float* b, __bf16* out, int C, long nVox) {
    hipMemsetAsync(stats, 0, (size_t)(2 * C + 1) * sizeof(float), stream);
    dim3 grid(64, C);
    bn_stats_k<<<grid, 256, 0, stream>>>(x, mask, stats, C, (int)nVox);
    long tot = nVox * C;
    bn_apply_k<<<(int)((tot + 255) / 256), 256, 0, stream>>>(
        x, mask, g, b, stats, C, tot, out);
  };

  auto concat = [&](const float* src, float* dst, long V, int C, int stride,
                    int off) {
    long tot = V * C;
    concat_copy_k<<<(int)((tot + 255) / 256), 256, 0, stream>>>(
        src, dst, V, C, stride, off);
  };

  // masks
  mask2float_k<<<(int)((V0 + 255) / 256), 256, 0, stream>>>(maskB, m0, V0);
  pool_mask_k<<<(int)((V1 + 255) / 256), 256, 0, stream>>>(m0, m1, Bn, S1, S1, S1);
  pool_mask_k<<<(int)((V2 + 255) / 256), 256, 0, stream>>>(m1, m2, Bn, S2, S2, S2);
  pool_mask_k<<<(int)((V3 + 255) / 256), 256, 0, stream>>>(m2, m3, Bn, S3, S3, S3);

  // layers_in: feats*m -> bf16, then sconv(w_in) -> 32ch @ L0
  feats2bf_k<<<(int)((V0 * 3 + 255) / 256), 256, 0, stream>>>(feats, m0, fbf,
                                                              V0, 3);
  conv(fbf, F(2), m0, nullptr, bufF, S0, S0, 3, 32, 3, 1, 1, 32, 0);

  // encoder level 0
  bn(bufF, m0, F(3), F(4), xbf, 32, V0);
  conv(xbf, F(5), m0, nullptr, dn0, S0, S0, 32, 32, 3, 1, 1, 32, 0);

  // level 1
  bn(dn0, m0, F(6), F(7), xbf, 32, V0);
  conv(xbf, F(8), m1, nullptr, bufF, S0, S1, 32, 64, 2, 2, 0, 64, 0);
  bn(bufF, m1, F(9), F(10), xbf, 64, V1);
  conv(xbf, F(11), m1, nullptr, dn1, S1, S1, 64, 64, 3, 1, 1, 64, 0);

  // level 2
  bn(dn1, m1, F(12), F(13), xbf, 64, V1);
  conv(xbf, F(14), m2, nullptr, bufF, S1, S2, 64, 96, 2, 2, 0, 96, 0);
  bn(bufF, m2, F(15), F(16), xbf, 96, V2);
  conv(xbf, F(17), m2, nullptr, dn2, S2, S2, 96, 96, 3, 1, 1, 96, 0);

  // level 3
  bn(dn2, m2, F(18), F(19), xbf, 96, V2);
  conv(xbf, F(20), m3, nullptr, bufF, S2, S3, 96, 128, 2, 2, 0, 128, 0);
  bn(bufF, m3, F(21), F(22), xbf, 128, V3);
  conv(xbf, F(23), m3, nullptr, dn3, S3, S3, 128, 128, 3, 1, 1, 128, 0);

  // decoder i0 (L3 -> L2, cp=96)
  bn(dn3, m3, F(24), F(25), xbf, 128, V3);
  upconv(xbf, F(26), m2, bufF, S3, S2, 128, 96, 192);
  concat(dn2, bufF, V2, 96, 192, 96);
  bn(bufF, m2, F(27), F(28), xbf, 192, V2);
  conv(xbf, F(29), m2, nullptr, bufF, S2, S2, 192, 96, 3, 1, 1, 96, 0);

  // decoder i1 (L2 -> L1, cp=64)
  bn(bufF, m2, F(30), F(31), xbf, 96, V2);
  upconv(xbf, F(32), m1, bufF, S2, S1, 96, 64, 128);
  concat(dn1, bufF, V1, 64, 128, 64);
  bn(bufF, m1, F(33), F(34), xbf, 128, V1);
  conv(xbf, F(35), m1, nullptr, bufF, S1, S1, 128, 64, 3, 1, 1, 64, 0);

  // decoder i2 (L1 -> L0, cp=32)
  bn(bufF, m1, F(36), F(37), xbf, 64, V1);
  upconv(xbf, F(38), m0, bufF, S1, S0, 64, 32, 64);
  concat(dn0, bufF, V0, 32, 64, 32);
  bn(bufF, m0, F(39), F(40), xbf, 64, V0);
  conv(xbf, F(41), m0, nullptr, bufF, S0, S0, 64, 32, 3, 1, 1, 32, 0);

  // layers_out: BN then linear (1^3 conv + bias) -> d_out
  bn(bufF, m0, F(42), F(43), xbf, 32, V0);
  conv(xbf, F(44), m0, F(45), (float*)d_out, S0, S0, 32, 20, 1, 1, 0, 20, 0);
}
